// QuadrupletInteraction_14714557956332
// MI455X (gfx1250) — matrix operand order repaired and checked by reading, not compiled, not run on hardware
//
#include <hip/hip_runtime.h>

// ---------------------------------------------------------------------------
// GemNet quadruplet interaction for MI455X (gfx1250), wave32 + WMMA f16.
// ---------------------------------------------------------------------------

typedef __attribute__((ext_vector_type(16))) _Float16 v16h;
typedef __attribute__((ext_vector_type(8)))  float    v8f;

#define E_TOT      200000
#define NB_        8
#define D_EDGE_    128
#define D_RBF_     16
#define D_CBF_     16
#define D_SBF_     32
#define D_QUAD_    64

#define INV_SQRT_2F   0.70710678118654752440f
#define INV_SQRT_NBF  0.35355339059327376220f

__device__ __forceinline__ float silu_f(float x) {
    return x / (1.0f + __expf(-x));
}

// --- WMMA wrapper: D = A(16x32 f16) * B(32x16 f16) + C(16x16 f32) ----------
__device__ __forceinline__ v8f wmma_f16(v16h a, v16h b, v8f c) {
    return __builtin_amdgcn_wmma_f32_16x16x32_f16(
        /*neg_a=*/false, a, /*neg_b=*/false, b,
        /*c_mod=*/(short)0, c, /*reuse_a=*/false, /*reuse_b=*/false);
}

// --- A-fragment: LDS tile row-major [16][ldk] f16, columns kofs..kofs+31 ---
// Per ISA 7.12.2 (16-bit A 16x32): lane L -> M=L&15, half=L>>4.
// VGPR v: K = kofs + (v>>2)*16 + half*8 + (v&3)*2   (pair K, K+1)
__device__ __forceinline__ v16h load_a_frag(const _Float16* tile, int ldk,
                                            int kofs, int lane) {
    int m  = lane & 15;
    int hs = lane >> 4;
    v16h a;
#pragma unroll
    for (int v = 0; v < 8; ++v) {
        int k = kofs + ((v >> 2) << 4) + (hs << 3) + ((v & 3) << 1);
        const _Float16* p = tile + m * ldk + k;
        a[2 * v]     = p[0];
        a[2 * v + 1] = p[1];
    }
    return a;
}

// --- B-fragment from transposed weight in LDS: wt[n][ldk] = W[k][n] --------
// lane L -> N = L&15 (col = ntile*16+N), VGPR v: K = (v>>2)*16 + (L>>4)*8 + (v&3)*2
__device__ __forceinline__ v16h load_b_frag(const _Float16* wt, int ldk,
                                            int ntile, int kofs, int lane) {
    int n  = lane & 15;
    int kb = (lane >> 4) << 3;
    const _Float16* p = wt + (ntile * 16 + n) * ldk + kofs + kb;
    v16h b;
#pragma unroll
    for (int v = 0; v < 8; ++v) {
        b[2 * v]     = p[2 * v + ((v >> 2) << 3)];
        b[2 * v + 1] = p[2 * v + 1 + ((v >> 2) << 3)];
    }
    return b;
}

// --- Block-cooperative weight load: global W[K][N] f32 -> LDS wt[N][ldk] f16
// (transposed, K-padded with zeros up to ldk). Coalesced global reads.
__device__ __forceinline__ void load_wt(_Float16* wt, const float* w,
                                        int K, int N, int ldk,
                                        int tid, int nthreads) {
    for (int i = tid; i < K * N; i += nthreads) {
        int k = i / N, n = i % N;
        wt[n * ldk + k] = (_Float16)w[i];
    }
    int pad = (ldk - K) * N;
    for (int i = tid; i < pad; i += nthreads) {
        int k = K + i / N, n = i % N;
        wt[n * ldk + k] = (_Float16)0.0f;
    }
}

__device__ __forceinline__ v8f zero_v8f() {
    v8f z;
#pragma unroll
    for (int v = 0; v < 8; ++v) z[v] = 0.0f;
    return z;
}

// ===========================================================================
// Kernel A: per-edge  h = silu(m_st@W_m_rbf) * (rbf@W_rbf) * s_rbf
//                     q = silu(h@W_m_cbf)              (stored f16, E x 64)
// block = 128 threads (4 waves), wave tile = 16 edges, grid = 3125
// ===========================================================================
__global__ void quad_kA(const float* __restrict__ m_st,
                        const float* __restrict__ rbf,
                        const float* __restrict__ sc_rbf,
                        const float* __restrict__ W_m_rbf,
                        const float* __restrict__ W_rbf,
                        const float* __restrict__ W_m_cbf,
                        _Float16*    __restrict__ q_out) {
    extern __shared__ __align__(16) char smem_raw[];
    _Float16* sm       = (_Float16*)smem_raw;
    _Float16* wt_mrbf  = sm;                    // [128][128] = 16384 h
    _Float16* wt_rbf   = wt_mrbf + 16384;       // [128][32]  =  4096 h (K pad)
    _Float16* wt_mcbf  = wt_rbf + 4096;         // [64][128]  =  8192 h
    _Float16* wstage   = wt_mcbf + 8192;        // 4 waves * 4608 h

    const int tid  = threadIdx.x;
    const int lane = tid & 31;
    const int wv   = tid >> 5;

    load_wt(wt_mrbf, W_m_rbf, 128, 128, 128, tid, blockDim.x);
    load_wt(wt_rbf,  W_rbf,    16, 128,  32, tid, blockDim.x);
    load_wt(wt_mcbf, W_m_cbf, 128,  64, 128, tid, blockDim.x);

    _Float16* st  = wstage + wv * 4608;   // m_st tile  [16][128]
    _Float16* hst = st + 2048;            // h tile     [16][128]
    _Float16* rA  = st + 4096;            // rbf tile   [16][32]

    const int tile = blockIdx.x * 4 + wv;
    const int e0   = tile * 16;

    // stage m_st tile (float4 coalesced)
    const float4* src = (const float4*)(m_st + (size_t)e0 * D_EDGE_);
    for (int i = lane; i < 512; i += 32) {
        float4 vv = src[i];
        int r = i >> 5, c = (i & 31) << 2;
        _Float16* d = st + r * 128 + c;
        d[0] = (_Float16)vv.x; d[1] = (_Float16)vv.y;
        d[2] = (_Float16)vv.z; d[3] = (_Float16)vv.w;
    }
    // stage rbf tile, K padded 16->32 with zeros
    for (int i = lane; i < 256; i += 32) {
        int r = i >> 4, c = i & 15;
        rA[r * 32 + c]      = (_Float16)rbf[(size_t)(e0 + r) * D_RBF_ + c];
        rA[r * 32 + 16 + c] = (_Float16)0.0f;
    }
    __syncthreads();

    const int n  = lane & 15;
    const int hs = lane >> 4;
    const float scr = sc_rbf[0];

    v16h a0 = load_a_frag(st, 128,  0, lane);
    v16h a1 = load_a_frag(st, 128, 32, lane);
    v16h a2 = load_a_frag(st, 128, 64, lane);
    v16h a3 = load_a_frag(st, 128, 96, lane);
    v16h ra = load_a_frag(rA,  32,  0, lane);

#pragma unroll
    for (int nt = 0; nt < 8; ++nt) {
        v8f acc = zero_v8f();
        acc = wmma_f16(a0, load_b_frag(wt_mrbf, 128, nt,  0, lane), acc);
        acc = wmma_f16(a1, load_b_frag(wt_mrbf, 128, nt, 32, lane), acc);
        acc = wmma_f16(a2, load_b_frag(wt_mrbf, 128, nt, 64, lane), acc);
        acc = wmma_f16(a3, load_b_frag(wt_mrbf, 128, nt, 96, lane), acc);
        v8f rc = zero_v8f();
        rc = wmma_f16(ra, load_b_frag(wt_rbf, 32, nt, 0, lane), rc);
#pragma unroll
        for (int v = 0; v < 8; ++v) {
            float hv = silu_f(acc[v]) * rc[v] * scr;
            hst[(hs * 8 + v) * 128 + nt * 16 + n] = (_Float16)hv;
        }
    }
    __syncthreads();

    v16h h0 = load_a_frag(hst, 128,  0, lane);
    v16h h1 = load_a_frag(hst, 128, 32, lane);
    v16h h2 = load_a_frag(hst, 128, 64, lane);
    v16h h3 = load_a_frag(hst, 128, 96, lane);

#pragma unroll
    for (int nt = 0; nt < 4; ++nt) {
        v8f acc = zero_v8f();
        acc = wmma_f16(h0, load_b_frag(wt_mcbf, 128, nt,  0, lane), acc);
        acc = wmma_f16(h1, load_b_frag(wt_mcbf, 128, nt, 32, lane), acc);
        acc = wmma_f16(h2, load_b_frag(wt_mcbf, 128, nt, 64, lane), acc);
        acc = wmma_f16(h3, load_b_frag(wt_mcbf, 128, nt, 96, lane), acc);
#pragma unroll
        for (int v = 0; v < 8; ++v) {
            q_out[(size_t)(e0 + hs * 8 + v) * D_QUAD_ + nt * 16 + n] =
                (_Float16)silu_f(acc[v]);
        }
    }
}

// ===========================================================================
// Kernel B: per (e,nb) row (1.6M rows):
//   g  = (q[b1]+q[b2]) * INV_SQRT_2
//   m  = g * (cbf@W_cbf) * s_cbf
//   r  = silu(m@W_m_sbf) * (sbf@W_sbf) * s_sbf
//   x[e] = sum_nb r * INV_SQRT_NB
// block = 256 threads (8 waves), wave tile = 16 rows (= 2 edges), grid 12500
// ===========================================================================
__global__ void quad_kB(const _Float16* __restrict__ q,
                        const float* __restrict__ cbf,
                        const float* __restrict__ sbf,
                        const int*   __restrict__ idx_s,
                        const int*   __restrict__ basis_idx1,
                        const int*   __restrict__ basis_idx2,
                        const float* __restrict__ W_cbf,
                        const float* __restrict__ W_m_sbf,
                        const float* __restrict__ W_sbf,
                        const float* __restrict__ sc_cbf,
                        const float* __restrict__ sc_sbf,
                        float*       __restrict__ x_out) {
    extern __shared__ __align__(16) char smem_raw[];
    _Float16* sm      = (_Float16*)smem_raw;
    _Float16* wt_cbf  = sm;                  // [64][32] = 2048 h (K pad 16->32)
    _Float16* wt_msbf = wt_cbf + 2048;       // [64][64] = 4096 h
    _Float16* wt_sbf  = wt_msbf + 4096;      // [64][32] = 2048 h
    float*    gall    = (float*)(smem_raw + 16384);          // 8 waves * 1024 f
    _Float16* hall    = (_Float16*)(smem_raw + 16384 + 32768); // 8 waves * 2048 h

    const int tid  = threadIdx.x;
    const int lane = tid & 31;
    const int wv   = tid >> 5;

    load_wt(wt_cbf,  W_cbf,  16, 64, 32, tid, blockDim.x);
    load_wt(wt_msbf, W_m_sbf, 64, 64, 64, tid, blockDim.x);
    load_wt(wt_sbf,  W_sbf,  32, 64, 32, tid, blockDim.x);

    float*    g  = gall + wv * 1024;            // [16][64] f32
    _Float16* cA = hall + wv * 2048;            // [16][32]
    _Float16* sA = cA + 512;                    // [16][32]
    _Float16* mA = sA + 512;                    // [16][64]

    const int tile = blockIdx.x * 8 + wv;
    const int base = tile * 16;                 // first flattened (e,nb) row
    const int e0   = base >> 3;                 // rows 0-7 -> e0, 8-15 -> e0+1

    // gather q[b1] + q[b2] into LDS (coalesced 128B row loads)
#pragma unroll 1
    for (int it = 0; it < 16; ++it) {
        int rid = base + it;
        int e   = rid >> 3;
        int nb  = rid & 7;
        int s   = idx_s[e];
        int i1  = basis_idx1[s * NB_ + nb];
        int i2  = basis_idx2[s * NB_ + nb];
        const _Float16* r1 = q + (size_t)i1 * D_QUAD_;
        const _Float16* r2 = q + (size_t)i2 * D_QUAD_;
        float u0 = (float)r1[lane * 2]     + (float)r2[lane * 2];
        float u1 = (float)r1[lane * 2 + 1] + (float)r2[lane * 2 + 1];
        g[it * 64 + lane * 2]     = u0 * INV_SQRT_2F;
        g[it * 64 + lane * 2 + 1] = u1 * INV_SQRT_2F;
    }
    // stage cbf (K pad 16->32) and sbf (K=32 exact)
    for (int i = lane; i < 256; i += 32) {
        int r = i >> 4, c = i & 15;
        cA[r * 32 + c]      = (_Float16)cbf[(size_t)(base + r) * D_CBF_ + c];
        cA[r * 32 + 16 + c] = (_Float16)0.0f;
    }
    for (int i = lane; i < 512; i += 32) {
        int r = i >> 5, c = i & 31;
        sA[r * 32 + c] = (_Float16)sbf[(size_t)(base + r) * D_SBF_ + c];
    }
    __syncthreads();

    const int n  = lane & 15;
    const int hs = lane >> 4;
    const float scc = sc_cbf[0];
    const float scs = sc_sbf[0];

    v16h ac = load_a_frag(cA, 32, 0, lane);
    v8f cacc[4];
#pragma unroll
    for (int nt = 0; nt < 4; ++nt) {
        cacc[nt] = zero_v8f();
        cacc[nt] = wmma_f16(ac, load_b_frag(wt_cbf, 32, nt, 0, lane), cacc[nt]);
    }
    // m = g * cW * s_cbf  ->  mA (A-operand layout) in LDS
#pragma unroll
    for (int nt = 0; nt < 4; ++nt) {
#pragma unroll
        for (int v = 0; v < 8; ++v) {
            float gv = g[(hs * 8 + v) * 64 + nt * 16 + n];
            mA[(hs * 8 + v) * 64 + nt * 16 + n] =
                (_Float16)(gv * cacc[nt][v] * scc);
        }
    }
    __syncthreads();

    v16h am0 = load_a_frag(mA, 64,  0, lane);
    v16h am1 = load_a_frag(mA, 64, 32, lane);
    v16h as  = load_a_frag(sA, 32,  0, lane);

#pragma unroll
    for (int nt = 0; nt < 4; ++nt) {
        v8f acc = zero_v8f();
        acc = wmma_f16(am0, load_b_frag(wt_msbf, 64, nt,  0, lane), acc);
        acc = wmma_f16(am1, load_b_frag(wt_msbf, 64, nt, 32, lane), acc);
        v8f sacc = zero_v8f();
        sacc = wmma_f16(as, load_b_frag(wt_sbf, 32, nt, 0, lane), sacc);
        // NB-reduction: rows 0-7 (lanes 0-15) are all 8 nb of edge e0,
        // rows 8-15 (lanes 16-31) are edge e0+1 -> sum accumulator VGPRs.
        float ssum = 0.0f;
#pragma unroll
        for (int v = 0; v < 8; ++v)
            ssum += silu_f(acc[v]) * sacc[v] * scs;
        x_out[(size_t)(e0 + hs) * D_QUAD_ + nt * 16 + n] = ssum * INV_SQRT_NBF;
    }
}

// ===========================================================================
// Kernel C: x128 = silu(x@W_dir); out = silu(x128@W_st); t = silu(x128@W_ts)
// block = 128 threads (4 waves), wave tile = 16 edges, grid = 3125
// ===========================================================================
__global__ void quad_kC(const float* __restrict__ x,
                        const float* __restrict__ W_dir,
                        const float* __restrict__ W_st,
                        const float* __restrict__ W_ts,
                        float*       __restrict__ out,
                        _Float16*    __restrict__ t_out) {
    extern __shared__ __align__(16) char smem_raw[];
    _Float16* sm     = (_Float16*)smem_raw;
    _Float16* wt_dir = sm;                  // [128][64]  =  8192 h
    _Float16* wt_st  = wt_dir + 8192;       // [128][128] = 16384 h
    _Float16* wt_ts  = wt_st + 16384;       // [128][128] = 16384 h
    _Float16* wstage = wt_ts + 16384;       // 4 waves * 3072 h

    const int tid  = threadIdx.x;
    const int lane = tid & 31;
    const int wv   = tid >> 5;

    load_wt(wt_dir, W_dir,  64, 128,  64, tid, blockDim.x);
    load_wt(wt_st,  W_st,  128, 128, 128, tid, blockDim.x);
    load_wt(wt_ts,  W_ts,  128, 128, 128, tid, blockDim.x);

    _Float16* xA  = wstage + wv * 3072;   // [16][64]
    _Float16* hst = xA + 1024;            // [16][128]

    const int tile = blockIdx.x * 4 + wv;
    const int e0   = tile * 16;

    for (int i = lane; i < 1024; i += 32) {
        int r = i >> 6, c = i & 63;
        xA[r * 64 + c] = (_Float16)x[(size_t)(e0 + r) * D_QUAD_ + c];
    }
    __syncthreads();

    const int n  = lane & 15;
    const int hs = lane >> 4;

    v16h xa0 = load_a_frag(xA, 64,  0, lane);
    v16h xa1 = load_a_frag(xA, 64, 32, lane);

#pragma unroll
    for (int nt = 0; nt < 8; ++nt) {
        v8f acc = zero_v8f();
        acc = wmma_f16(xa0, load_b_frag(wt_dir, 64, nt,  0, lane), acc);
        acc = wmma_f16(xa1, load_b_frag(wt_dir, 64, nt, 32, lane), acc);
#pragma unroll
        for (int v = 0; v < 8; ++v)
            hst[(hs * 8 + v) * 128 + nt * 16 + n] = (_Float16)silu_f(acc[v]);
    }
    __syncthreads();

    v16h h0 = load_a_frag(hst, 128,  0, lane);
    v16h h1 = load_a_frag(hst, 128, 32, lane);
    v16h h2 = load_a_frag(hst, 128, 64, lane);
    v16h h3 = load_a_frag(hst, 128, 96, lane);

#pragma unroll
    for (int nt = 0; nt < 8; ++nt) {
        v8f acc = zero_v8f();
        acc = wmma_f16(h0, load_b_frag(wt_st, 128, nt,  0, lane), acc);
        acc = wmma_f16(h1, load_b_frag(wt_st, 128, nt, 32, lane), acc);
        acc = wmma_f16(h2, load_b_frag(wt_st, 128, nt, 64, lane), acc);
        acc = wmma_f16(h3, load_b_frag(wt_st, 128, nt, 96, lane), acc);
#pragma unroll
        for (int v = 0; v < 8; ++v)
            out[(size_t)(e0 + hs * 8 + v) * D_EDGE_ + nt * 16 + n] =
                silu_f(acc[v]);
    }
#pragma unroll
    for (int nt = 0; nt < 8; ++nt) {
        v8f acc = zero_v8f();
        acc = wmma_f16(h0, load_b_frag(wt_ts, 128, nt,  0, lane), acc);
        acc = wmma_f16(h1, load_b_frag(wt_ts, 128, nt, 32, lane), acc);
        acc = wmma_f16(h2, load_b_frag(wt_ts, 128, nt, 64, lane), acc);
        acc = wmma_f16(h3, load_b_frag(wt_ts, 128, nt, 96, lane), acc);
#pragma unroll
        for (int v = 0; v < 8; ++v)
            t_out[(size_t)(e0 + hs * 8 + v) * D_EDGE_ + nt * 16 + n] =
                (_Float16)silu_f(acc[v]);
    }
}

// ===========================================================================
// Kernel D: out[e] = (x_st[e] + t[idx_swap[e]]) * INV_SQRT_2
// ===========================================================================
__global__ void quad_kD(float* __restrict__ out,
                        const _Float16* __restrict__ t,
                        const int* __restrict__ idx_swap) {
    int i = blockIdx.x * blockDim.x + threadIdx.x;
    if (i < E_TOT * D_EDGE_) {
        int e = i >> 7;
        int c = i & 127;
        int es = idx_swap[e];
        float tv = (float)t[(size_t)es * D_EDGE_ + c];
        out[i] = (out[i] + tv) * INV_SQRT_2F;
    }
}

// ===========================================================================
extern "C" void kernel_launch(void* const* d_in, const int* in_sizes, int n_in,
                              void* d_out, int out_size, void* d_ws,
                              size_t ws_size, hipStream_t stream) {
    (void)in_sizes; (void)n_in; (void)out_size; (void)ws_size;

    const float* m_st     = (const float*)d_in[0];
    const float* rbf      = (const float*)d_in[1];
    const float* cbf      = (const float*)d_in[2];
    const float* sbf      = (const float*)d_in[3];
    const int*   idx_s    = (const int*)d_in[4];
    const int*   idx_swap = (const int*)d_in[5];
    const int*   b1       = (const int*)d_in[6];
    const int*   b2       = (const int*)d_in[7];
    const float* W_m_rbf  = (const float*)d_in[8];
    const float* W_rbf    = (const float*)d_in[9];
    const float* W_m_cbf  = (const float*)d_in[10];
    const float* W_cbf    = (const float*)d_in[11];
    const float* W_m_sbf  = (const float*)d_in[12];
    const float* W_sbf    = (const float*)d_in[13];
    const float* W_dir    = (const float*)d_in[14];
    const float* W_st     = (const float*)d_in[15];
    const float* W_ts     = (const float*)d_in[16];
    const float* sc_rbf   = (const float*)d_in[17];
    const float* sc_cbf   = (const float*)d_in[18];
    const float* sc_sbf   = (const float*)d_in[19];

    float* out = (float*)d_out;

    // Workspace layout (q dead after kB, so t overlaps q):
    //   x : [0, 51.2MB)               E*64 f32
    //   q : [51.2MB, 76.8MB)          E*64 f16   (L2-resident gather table)
    //   t : [51.2MB, 102.4MB)         E*128 f16  (written in kC, after kB)
    char* ws = (char*)d_ws;
    float*    x_ws = (float*)ws;
    _Float16* q_ws = (_Float16*)(ws + (size_t)E_TOT * D_QUAD_ * 4);
    _Float16* t_ws = (_Float16*)(ws + (size_t)E_TOT * D_QUAD_ * 4);

    // Kernel A: 12500 edge-tiles / 4 waves per block = 3125 blocks
    size_t smemA = (size_t)(16384 + 4096 + 8192 + 4 * 4608) * 2;
    quad_kA<<<3125, 128, smemA, stream>>>(m_st, rbf, sc_rbf, W_m_rbf, W_rbf,
                                          W_m_cbf, q_ws);

    // Kernel B: 100000 row-tiles / 8 waves per block = 12500 blocks
    size_t smemB = 16384 + 32768 + 8 * 2048 * 2;
    quad_kB<<<12500, 256, smemB, stream>>>(q_ws, cbf, sbf, idx_s, b1, b2,
                                           W_cbf, W_m_sbf, W_sbf,
                                           sc_cbf, sc_sbf, x_ws);

    // Kernel C: 3125 blocks of 4 waves
    size_t smemC = (size_t)(8192 + 16384 + 16384 + 4 * 3072) * 2;
    quad_kC<<<3125, 128, smemC, stream>>>(x_ws, W_dir, W_st, W_ts, out, t_ws);

    // Kernel D: swap-gather combine
    quad_kD<<<(E_TOT * D_EDGE_ + 255) / 256, 256, 0, stream>>>(out, t_ws,
                                                               idx_swap);
}
